// MultiHeadFAVORAttention_9534827397489
// MI455X (gfx1250) — compile-verified, matrix operand order checked by reading, not run on hardware
//
#include <hip/hip_runtime.h>
#include <hip/hip_bf16.h>
#include <math.h>

typedef _Float16 half_t;
typedef __attribute__((ext_vector_type(16))) _Float16 v16h;
typedef __attribute__((ext_vector_type(8)))  _Float16 h8;
typedef __attribute__((ext_vector_type(8)))  float    v8f;

#define B_    8
#define S_    4096
#define D_    768
#define H_    8
#define HD_   96
#define M_    256
#define G_    3
#define R_    128
#define GM_   768          // G_*M_
#define NF_   2304         // 3*GM_
#define ROWS_ 32768        // B_*S_
#define BH_   64           // B_*H_
#define BHS_  262144       // BH_*S_

#define TM_   128          // block tile M
#define TN_   64           // block tile N
#define KS_   32           // K per stage
#define PAD_  40           // padded LDS row stride in halves (32 + 8)

enum { EPI_COS = 0, EPI_EXPQ, EPI_EXPKT, EPI_TF16, EPI_DIV, EPI_BIASF32 };

// ---------------------------------------------------------------------------
// Shared epilogue
// ---------------------------------------------------------------------------
template <int EPI>
__device__ inline void epi_write(float v, long m, int n, int bh,
                                 half_t* __restrict__ outH,
                                 float* __restrict__ outF, long sOut, int ldc,
                                 const float* __restrict__ p1,
                                 const float* __restrict__ p2) {
  if constexpr (EPI == EPI_COS) {
    outH[(size_t)m * ldc + n] =
        (half_t)(0.08838834764831845f * cosf(v + p1[n]));
  } else if constexpr (EPI == EPI_EXPQ) {
    outH[(size_t)m * ldc + n] = (half_t)__expf(v + p1[n] - 0.5f * p2[m]);
  } else if constexpr (EPI == EPI_EXPKT) {
    float e = __expf(v + p1[n] - 0.5f * p2[m]);
    const long bhr = m >> 12;                       // row = bh*4096 + s
    const int  s   = (int)(m & 4095);
    outH[((size_t)(bhr * R_ + n)) * S_ + s] = (half_t)e;          // kp^T
  } else if constexpr (EPI == EPI_TF16) {
    (outH + (long)bh * sOut)[(size_t)n * ldc + m] = (half_t)v;    // kv^T
  } else if constexpr (EPI == EPI_DIV) {
    const int b = bh >> 3, h = bh & 7;
    const float zz = p2[(size_t)bh * S_ + m];
    outH[((size_t)(b * S_ + m)) * D_ + h * HD_ + n] =
        (half_t)(v / (zz + 1e-6f));                 // merged-head scatter
  } else {  // EPI_BIASF32
    outF[(size_t)m * ldc + n] = v + p1[n];
  }
}

// ---------------------------------------------------------------------------
// Fragment loads (CDNA5 16-bit A layout; B packed [N][K] uses mirror layout)
// ---------------------------------------------------------------------------
__device__ inline v16h frag_from(const half_t* __restrict__ p0) {
  h8 c0 = *(const h8*)(p0);
  h8 c1 = *(const h8*)(p0 + 16);
  v16h a;
#pragma unroll
  for (int i = 0; i < 8; ++i) { a[i] = c0[i]; a[8 + i] = c1[i]; }
  return a;
}

__device__ inline v16h load_frag_g(const half_t* __restrict__ base, int ld,
                                   int rbase, int k0, int lane) {
  const int r = lane & 15;
  const int hi = lane >> 4;
  return frag_from(base + (size_t)(rbase + r) * ld + k0 + hi * 8);
}

__device__ inline float wave_red(float v) {
#pragma unroll
  for (int o = 16; o > 0; o >>= 1) v += __shfl_down(v, o, 32);
  return v;
}

// Async global -> LDS 16B copy (per lane), tracked by ASYNCcnt.
__device__ inline void async_cp16(unsigned lds_byte_off, const half_t* g) {
  asm volatile("global_load_async_to_lds_b128 %0, %1, off"
               :: "v"(lds_byte_off), "v"(g)
               : "memory");
}
__device__ inline void wait_async0() {
  asm volatile("s_wait_asynccnt 0x0" ::: "memory");
}
__device__ inline unsigned lds_off(const void* p) {
  return (unsigned)(uintptr_t)p;   // low 32 bits of shared aperture = LDS offset
}

// ---------------------------------------------------------------------------
// Tiled WMMA GEMM with async-LDS double buffering.
//   C[M,N] = A[M,K] * Bt[N,K]^T  (f16 in, f32 acc)
//   block tile 128x64, wave tile 32x32 (2x2 wmma), 8 waves (4x2).
//   Requires M%128==0, N%64==0, K%32==0 (all call sites satisfy this).
// ---------------------------------------------------------------------------
template <int EPI>
__global__ __launch_bounds__(256) void gemm_tiled(
    const half_t* __restrict__ A, int lda,
    const half_t* __restrict__ Bt, int ldb, int Kdim,
    half_t* __restrict__ outH, float* __restrict__ outF, int ldc,
    const float* __restrict__ p1, const float* __restrict__ p2) {
  __shared__ __align__(16) half_t shA[2][TM_ * PAD_];
  __shared__ __align__(16) half_t shB[2][TN_ * PAD_];

  const int tid  = threadIdx.x;
  const int wave = tid >> 5;
  const int lane = tid & 31;
  const int wm   = wave & 3;          // 0..3  -> M offset wm*32
  const int wn   = wave >> 2;         // 0..1  -> N offset wn*32
  const long blockM = (long)blockIdx.x * TM_;
  const long blockN = (long)blockIdx.y * TN_;
  const int  nk = Kdim / KS_;

  // per-thread copy assignments (16B chunks)
  const int rA0 = tid >> 2, kcA = tid & 3;          // A chunk 0: rows 0..63
  const int rA1 = rA0 + 64;                          // A chunk 1: rows 64..127
  const int rB  = tid >> 2, kcB = tid & 3;          // B: rows 0..63

#define ISSUE_STAGE(buf, t)                                                   \
  do {                                                                        \
    const int k0 = (t) * KS_;                                                 \
    async_cp16(lds_off(&shA[(buf)][rA0 * PAD_ + kcA * 8]),                    \
               A + (blockM + rA0) * (size_t)lda + k0 + kcA * 8);              \
    async_cp16(lds_off(&shA[(buf)][rA1 * PAD_ + kcA * 8]),                    \
               A + (blockM + rA1) * (size_t)lda + k0 + kcA * 8);              \
    async_cp16(lds_off(&shB[(buf)][rB * PAD_ + kcB * 8]),                     \
               Bt + (blockN + rB) * (size_t)ldb + k0 + kcB * 8);              \
  } while (0)

  v8f acc[2][2];
#pragma unroll
  for (int i = 0; i < 2; ++i)
#pragma unroll
    for (int j = 0; j < 2; ++j) acc[i][j] = (v8f){0.f,0.f,0.f,0.f,0.f,0.f,0.f,0.f};

  ISSUE_STAGE(0, 0);

  const int fr = lane & 15, fhi = (lane >> 4) * 8;
  for (int t = 0; t < nk; ++t) {
    const int buf = t & 1;
    wait_async0();
    __syncthreads();                      // stage t resident in LDS
    if (t + 1 < nk) ISSUE_STAGE(buf ^ 1, t + 1);

    v16h af[2], bf[2];
#pragma unroll
    for (int fi = 0; fi < 2; ++fi)
      af[fi] = frag_from(&shA[buf][(wm * 32 + fi * 16 + fr) * PAD_ + fhi]);
#pragma unroll
    for (int fj = 0; fj < 2; ++fj)
      bf[fj] = frag_from(&shB[buf][(wn * 32 + fj * 16 + fr) * PAD_ + fhi]);
#pragma unroll
    for (int fi = 0; fi < 2; ++fi)
#pragma unroll
      for (int fj = 0; fj < 2; ++fj)
        acc[fi][fj] = __builtin_amdgcn_wmma_f32_16x16x32_f16(
            false, af[fi], false, bf[fj], (short)0, acc[fi][fj], false, false);

    __syncthreads();                      // protect buffer reuse (WAR)
  }
#undef ISSUE_STAGE

  const int n_loc = lane & 15;
  const int mbase = (lane >> 4) * 8;
#pragma unroll
  for (int fi = 0; fi < 2; ++fi) {
#pragma unroll
    for (int fj = 0; fj < 2; ++fj) {
      const long mrow = blockM + wm * 32 + fi * 16;
      const int  ncol = (int)blockN + wn * 32 + fj * 16;
#pragma unroll
      for (int j = 0; j < 8; ++j) {
        epi_write<EPI>(acc[fi][fj][j], mrow + mbase + j, ncol + n_loc, 0,
                       outH, outF, 0, ldc, p1, p2);
      }
    }
  }
}

// ---------------------------------------------------------------------------
// Wave-per-16x16-tile GEMM (direct global loads) for small / odd shapes.
// ---------------------------------------------------------------------------
template <int EPI>
__global__ __launch_bounds__(256) void gemm16(
    const half_t* __restrict__ A,  int lda, long sA,
    const half_t* __restrict__ Bt, int ldb, long sB,
    int Mdim, int Ndim, int Kdim,
    half_t* __restrict__ outH, float* __restrict__ outF, long sOut, int ldc,
    const float* __restrict__ p1, const float* __restrict__ p2) {
  const int wave = threadIdx.x >> 5;
  const int lane = threadIdx.x & 31;
  const int tilesN = Ndim >> 4;
  const long tile = (long)blockIdx.x * 8 + wave;
  const long totalTiles = (long)(Mdim >> 4) * tilesN;
  if (tile >= totalTiles) return;
  const int tm = (int)(tile / tilesN);
  const int tn = (int)(tile % tilesN);
  const int bh = blockIdx.z;
  const half_t* Ab = A  + (long)bh * sA;
  const half_t* Bb = Bt + (long)bh * sB;
  const int mrow = tm << 4;
  const int ncol = tn << 4;

  v8f acc = {0.f, 0.f, 0.f, 0.f, 0.f, 0.f, 0.f, 0.f};
  for (int k0 = 0; k0 < Kdim; k0 += 32) {
    if (k0 + 64 <= Kdim) {
      __builtin_prefetch(Ab + (size_t)(mrow + (lane & 15)) * lda + k0 + 32, 0, 1);
      __builtin_prefetch(Bb + (size_t)(ncol + (lane & 15)) * ldb + k0 + 32, 0, 1);
    }
    v16h a = load_frag_g(Ab, lda, mrow, k0, lane);
    v16h b = load_frag_g(Bb, ldb, ncol, k0, lane);
    acc = __builtin_amdgcn_wmma_f32_16x16x32_f16(false, a, false, b,
                                                 (short)0, acc, false, false);
  }
  const int n_loc = lane & 15;
  const int mbase = (lane >> 4) * 8;
#pragma unroll
  for (int j = 0; j < 8; ++j) {
    epi_write<EPI>(acc[j], (long)(mrow + mbase + j), ncol + n_loc, bh,
                   outH, outF, sOut, ldc, p1, p2);
  }
}

// ---------------------------------------------------------------------------
// Packing / elementwise helpers
// ---------------------------------------------------------------------------
__global__ __launch_bounds__(256) void pack_xh(const float* __restrict__ x,
                                               half_t* __restrict__ Xh, long n) {
  long i = (long)blockIdx.x * 256 + threadIdx.x;
  if (i < n) Xh[i] = (half_t)x[i];
}

__global__ __launch_bounds__(256) void pack_w(
    const float* __restrict__ Wq, const float* __restrict__ Wk,
    const float* __restrict__ Wv, const float* __restrict__ bq,
    const float* __restrict__ bk, const float* __restrict__ bv,
    half_t* __restrict__ WhT, float* __restrict__ bAll) {
  long i = (long)blockIdx.x * 256 + threadIdx.x;   // [NF_][D_]
  if (i >= (long)NF_ * D_) return;
  const int n = (int)(i / D_);
  const int d = (int)(i % D_);
  const int blk = n / GM_;                          // 0:q 1:k 2:v
  const int nn  = n % GM_;
  const int g = nn / M_, m = nn % M_;
  const float* W = (blk == 0) ? Wq : (blk == 1) ? Wk : Wv;
  WhT[i] = (half_t)W[((long)g * D_ + d) * M_ + m];
  if (d == 0) {
    const float* bb = (blk == 0) ? bq : (blk == 1) ? bk : bv;
    bAll[n] = bb[g * M_ + m];
  }
}

__global__ __launch_bounds__(256) void pack_fw(const float* __restrict__ fW,
                                               half_t* __restrict__ fWT) {
  int i = blockIdx.x * 256 + threadIdx.x;           // [R_][HD_]
  if (i >= R_ * HD_) return;
  int r = i / HD_, d = i % HD_;
  fWT[i] = (half_t)fW[(long)d * R_ + r];
}

__global__ __launch_bounds__(256) void pack_ow(const float* __restrict__ oW,
                                               half_t* __restrict__ oWT) {
  long i = (long)blockIdx.x * 256 + threadIdx.x;    // [D_][D_] transposed
  if (i >= (long)D_ * D_) return;
  int n = (int)(i / D_), k = (int)(i % D_);
  oWT[i] = (half_t)oW[(long)k * D_ + n];
}

// Rotary (one wave per head-row; q OR k) + squared-norm reduce.
__global__ __launch_bounds__(256) void rotary_kernel(
    const half_t* __restrict__ F, half_t* __restrict__ Qrot,
    half_t* __restrict__ Krot, float* __restrict__ nsqQ,
    float* __restrict__ nsqK) {
  const long wid = (long)blockIdx.x * 8 + (threadIdx.x >> 5);
  const int lane = threadIdx.x & 31;
  if (wid >= 2L * BHS_) return;
  const bool isK = wid >= (long)BHS_;
  const long row = isK ? wid - BHS_ : wid;          // (b*H + h)*S + s
  const long bhr = row >> 12;
  const int  s   = (int)(row & 4095);
  const long b   = bhr >> 3;
  const int  h   = (int)(bhr & 7);
  const half_t* src = F + ((size_t)(b * S_ + s)) * NF_ + (isK ? GM_ : 0) + h * HD_;
  half_t* dst = (isK ? Krot : Qrot) + (size_t)row * HD_;
  const float logB = logf(12315.0f);
  float nsq = 0.f;
#pragma unroll
  for (int t = 0; t < 2; ++t) {
    const int i = lane + t * 32;                    // rotary pair index (0..47)
    if (i < 48) {
      const float x1 = (float)src[i];
      const float x2 = (float)src[48 + i];
      const float invf = __expf(-(2.0f * i / 96.0f) * logB);
      float sn, cs;
      __sincosf((float)s * invf, &sn, &cs);
      dst[i]      = (half_t)(x1 * cs - x2 * sn);
      dst[48 + i] = (half_t)(x1 * sn + x2 * cs);
      nsq += x1 * x1 + x2 * x2;                     // rotation preserves norm
    }
  }
  nsq = wave_red(nsq);
  if (lane == 0) (isK ? nsqK : nsqQ)[row] = nsq;
}

// ksum[bh*R + r] = sum_s kpT[bh*R + r][s]
__global__ __launch_bounds__(256) void ksum_kernel(const half_t* __restrict__ kpT,
                                                   float* __restrict__ ksum) {
  const int row = blockIdx.x;                        // BH_*R_ rows
  const half_t* p = kpT + (size_t)row * S_;
  float a = 0.f;
  for (int i = threadIdx.x; i < S_; i += 256) a += (float)p[i];
  __shared__ float sm[256];
  sm[threadIdx.x] = a;
  __syncthreads();
  for (int s = 128; s > 0; s >>= 1) {
    if (threadIdx.x < s) sm[threadIdx.x] += sm[threadIdx.x + s];
    __syncthreads();
  }
  if (threadIdx.x == 0) ksum[row] = sm[0];
}

// z[row] = qp[row,:] . ksum[bh,:]
__global__ __launch_bounds__(256) void z_kernel(const half_t* __restrict__ qp,
                                                const float* __restrict__ ksum,
                                                float* __restrict__ z) {
  const long wid = (long)blockIdx.x * 8 + (threadIdx.x >> 5);
  const int lane = threadIdx.x & 31;
  if (wid >= (long)BHS_) return;
  const half_t* q = qp + (size_t)wid * R_;
  const float*  k = ksum + (size_t)(wid >> 12) * R_;
  float a = 0.f;
#pragma unroll
  for (int i = lane; i < R_; i += 32) a += (float)q[i] * k[i];
  a = wave_red(a);
  if (lane == 0) z[wid] = a;
}

// VT[(bh*HD + d)][s] = F[(b*S + s)][2*GM + h*HD + d]
__global__ __launch_bounds__(256) void vtrans_kernel(const half_t* __restrict__ F,
                                                     half_t* __restrict__ VT) {
  long i = (long)blockIdx.x * 256 + threadIdx.x;
  if (i >= (long)BH_ * HD_ * S_) return;
  const int  s  = (int)(i & 4095);
  const long rd = i >> 12;                           // bh*HD + d
  const long bh = rd / HD_;
  const int  d  = (int)(rd % HD_);
  const long b  = bh >> 3;
  const int  h  = (int)(bh & 7);
  VT[i] = F[((size_t)(b * S_ + s)) * NF_ + 2 * GM_ + h * HD_ + d];
}

// ---------------------------------------------------------------------------
extern "C" void kernel_launch(void* const* d_in, const int* in_sizes, int n_in,
                              void* d_out, int out_size, void* d_ws,
                              size_t ws_size, hipStream_t stream) {
  (void)in_sizes; (void)n_in; (void)out_size; (void)ws_size;
  const float* x  = (const float*)d_in[0];
  const float* Wq = (const float*)d_in[1];
  const float* bq = (const float*)d_in[2];
  const float* Wk = (const float*)d_in[3];
  const float* bk = (const float*)d_in[4];
  const float* Wv = (const float*)d_in[5];
  const float* bv = (const float*)d_in[6];
  const float* fW = (const float*)d_in[7];
  const float* fb = (const float*)d_in[8];
  const float* oW = (const float*)d_in[9];
  const float* ob = (const float*)d_in[10];
  float* out = (float*)d_out;

  char* ws = (char*)d_ws;
  size_t o = 0;
  auto alloc = [&](size_t bytes) -> char* {
    char* p = ws + o;
    o += (bytes + 255) & ~(size_t)255;
    return p;
  };
  half_t* Xh   = (half_t*)alloc((size_t)ROWS_ * D_ * 2);
  half_t* WhT  = (half_t*)alloc((size_t)NF_ * D_ * 2);
  float*  bAll = (float*) alloc((size_t)NF_ * 4);
  half_t* fWT  = (half_t*)alloc((size_t)R_ * HD_ * 2);
  half_t* oWT  = (half_t*)alloc((size_t)D_ * D_ * 2);
  half_t* F    = (half_t*)alloc((size_t)ROWS_ * NF_ * 2);
  half_t* Qrot = (half_t*)alloc((size_t)BHS_ * HD_ * 2);   // later reused: Attn
  half_t* Krot = (half_t*)alloc((size_t)BHS_ * HD_ * 2);   // later reused: VT
  float*  nsqQ = (float*) alloc((size_t)BHS_ * 4);
  float*  nsqK = (float*) alloc((size_t)BHS_ * 4);
  half_t* qp   = (half_t*)alloc((size_t)BHS_ * R_ * 2);
  half_t* kpT  = (half_t*)alloc((size_t)BHS_ * R_ * 2);
  float*  ksum = (float*) alloc((size_t)BH_ * R_ * 4);
  float*  z    = (float*) alloc((size_t)BHS_ * 4);
  half_t* kvT  = (half_t*)alloc((size_t)BH_ * HD_ * R_ * 2);
  half_t* Attn = Qrot;   // Qrot dead after FAVOR-q GEMM
  half_t* VT   = Krot;   // Krot dead after FAVOR-k GEMM

  // 1) pack inputs to f16 (A row-major, B as [N][K])
  pack_xh<<<(unsigned)(((long)ROWS_ * D_ + 255) / 256), 256, 0, stream>>>(
      x, Xh, (long)ROWS_ * D_);
  pack_w<<<(unsigned)(((long)NF_ * D_ + 255) / 256), 256, 0, stream>>>(
      Wq, Wk, Wv, bq, bk, bv, WhT, bAll);
  pack_fw<<<(R_ * HD_ + 255) / 256, 256, 0, stream>>>(fW, fWT);
  pack_ow<<<(unsigned)(((long)D_ * D_ + 255) / 256), 256, 0, stream>>>(oW, oWT);

  // 2) multiscale features (q|k|v fused): F = sqrt(2/M)*cos(X*W + b)
  gemm_tiled<EPI_COS><<<dim3(ROWS_ / TM_, NF_ / TN_, 1), 256, 0, stream>>>(
      Xh, D_, WhT, D_, D_, F, nullptr, NF_, bAll, nullptr);

  // 3) rotary + squared norms for q and k heads
  rotary_kernel<<<(unsigned)((2L * BHS_ + 7) / 8), 256, 0, stream>>>(
      F, Qrot, Krot, nsqQ, nsqK);

  // 4) FAVOR projections: qp (row-major) and kp^T
  gemm_tiled<EPI_EXPQ><<<dim3(BHS_ / TM_, R_ / TN_, 1), 256, 0, stream>>>(
      Qrot, HD_, fWT, HD_, HD_, qp, nullptr, R_, fb, nsqQ);
  gemm_tiled<EPI_EXPKT><<<dim3(BHS_ / TM_, R_ / TN_, 1), 256, 0, stream>>>(
      Krot, HD_, fWT, HD_, HD_, kpT, nullptr, R_, fb, nsqK);

  // 5) v^T per head (B operand for kv GEMM); Krot buffer is now free
  vtrans_kernel<<<(unsigned)(((long)BH_ * HD_ * S_ + 255) / 256), 256, 0,
                  stream>>>(F, VT);

  // 6) ksum and z normalizer
  ksum_kernel<<<BH_ * R_, 256, 0, stream>>>(kpT, ksum);
  z_kernel<<<(unsigned)((BHS_ + 7) / 8), 256, 0, stream>>>(qp, ksum, z);

  // 7) kv^T = (kp^T * v)^T per (b,h): batched, K = S = 4096
  {
    long tiles = (long)(R_ / 16) * (HD_ / 16);      // 48 per batch
    gemm16<EPI_TF16><<<dim3((unsigned)((tiles + 7) / 8), 1, BH_), 256, 0,
                       stream>>>(
        kpT, S_, (long)R_ * S_, VT, S_, (long)HD_ * S_, R_, HD_, S_,
        kvT, nullptr, (long)HD_ * R_, R_, nullptr, nullptr);
  }

  // 8) out = (qp * kv) / (z + eps), merged-head scatter into Attn (f16)
  {
    long tiles = (long)(S_ / 16) * (HD_ / 16);      // 1536 per batch
    gemm16<EPI_DIV><<<dim3((unsigned)((tiles + 7) / 8), 1, BH_), 256, 0,
                      stream>>>(
        qp, R_, (long)S_ * R_, kvT, R_, (long)HD_ * R_, S_, HD_, R_,
        Attn, nullptr, 0, 0, nullptr, z);
  }

  // 9) final projection: out = Attn * out_W + out_b (f32)
  gemm_tiled<EPI_BIASF32><<<dim3(ROWS_ / TM_, D_ / TN_, 1), 256, 0, stream>>>(
      Attn, D_, oWT, D_, D_, nullptr, out, D_, ob, nullptr);
}
